// InterventionOnline_88459146428728
// MI455X (gfx1250) — matrix operand order repaired and checked by reading, not compiled, not run on hardware
//
#include <hip/hip_runtime.h>
#include <math.h>

typedef __attribute__((ext_vector_type(16))) __bf16   bf16x16;
typedef __attribute__((ext_vector_type(8)))  __bf16   bf16x8;
typedef __attribute__((ext_vector_type(16))) _Float16 f16x16;
typedef __attribute__((ext_vector_type(8)))  _Float16 f16x8;
typedef __attribute__((ext_vector_type(8)))  float    v8f;

constexpr int N = 8192, M = 8192, D = 512;
constexpr int NCHUNK = 2048;   // S chunk rows: 2048*8192*2B = 32MB, L2 resident
constexpr int MSPLIT = 8;      // parallel m-splits for softmax stats pass
constexpr int MSTAGE = 64;     // k rows staged in LDS per block iteration (64KB)
constexpr float RSCALE = 0.04419417382415922f; // 1/sqrt(512)

// ---------------- WMMA wrappers ----------------
__device__ inline v8f wmma_bf16(bf16x16 a, bf16x16 b, v8f c) {
  return __builtin_amdgcn_wmma_f32_16x16x32_bf16(false, a, false, b, (short)0, c, false, false);
}
__device__ inline v8f wmma_f16(f16x16 a, f16x16 b, v8f c) {
  return __builtin_amdgcn_wmma_f32_16x16x32_f16(false, a, false, b, (short)0, c, false, false);
}
__device__ inline v8f vzero8() { v8f z = {0.f,0.f,0.f,0.f,0.f,0.f,0.f,0.f}; return z; }

// ---------------- fragment loads (NT layout, contiguous k per lane half, same order in A and B) ----------------
// A 16x32 bf16: lanes 0-15 = row m, K {0..15}; lanes 16-31 = row m, K {16..31}
__device__ inline bf16x16 load_a_bf16(const __bf16* A, int lda, int mBase, int kBase, int lane) {
  const int m   = mBase + (lane & 15);
  const int klo = kBase + ((lane >> 4) << 4);
  const __bf16* p = A + (size_t)m * lda + klo;
  union { bf16x16 v; bf16x8 h[2]; } u;
  u.h[0] = *(const bf16x8*)p;
  u.h[1] = *(const bf16x8*)(p + 8);
  return u.v;
}
// B 32x16 bf16 (NT: "column" n is a K-contiguous row of the second operand)
__device__ inline bf16x16 load_b_bf16(const __bf16* B, int ldb, int nBase, int kBase, int lane) {
  const int n   = nBase + (lane & 15);
  const int klo = kBase + ((lane >> 4) << 4);
  return *(const bf16x16*)(B + (size_t)n * ldb + klo);
}
__device__ inline f16x16 load_b_f16(const _Float16* B, int ldb, int nBase, int kBase, int lane) {
  const int n   = nBase + (lane & 15);
  const int klo = kBase + ((lane >> 4) << 4);
  return *(const f16x16*)(B + (size_t)n * ldb + klo);
}
// A-frag from S with fused softmax normalization: p = exp(s - max) * inv_sum
__device__ inline f16x16 load_p_f16(const _Float16* S, int lds_, int rowBase, int kBase,
                                    int lane, float mx, float is) {
  const int row = rowBase + (lane & 15);
  const int klo = kBase + ((lane >> 4) << 4);
  const _Float16* p = S + (size_t)row * lds_ + klo;
  union { f16x8 v; _Float16 e[8]; } lo, hi2;
  lo.v  = *(const f16x8*)p;
  hi2.v = *(const f16x8*)(p + 8);
  union { f16x16 v; _Float16 e[16]; } u;
#pragma unroll
  for (int i = 0; i < 8; ++i) {
    u.e[i]     = (_Float16)(__expf((float)lo.e[i]  - mx) * is);
    u.e[i + 8] = (_Float16)(__expf((float)hi2.e[i] - mx) * is);
  }
  return u.v;
}
// C/D 16x16 f32: VGPR r, lanes 0-15 -> row r, lanes 16-31 -> row 8+r; col = lane&15
__device__ inline void store_tile(__bf16* outB, _Float16* outT, int ldo, int ldt,
                                  int mBase, int nBase, int lane, v8f acc, const float* bias) {
  const int hi = lane >> 4, col = nBase + (lane & 15);
  const float bv = bias ? bias[col] : 0.0f;
#pragma unroll
  for (int r = 0; r < 8; ++r) {
    const int row = mBase + hi * 8 + r;
    const float v = acc[r] + bv;
    outB[(size_t)row * ldo + col] = (__bf16)v;
    if (outT) outT[(size_t)col * ldt + row] = (_Float16)v;  // transposed f16 copy (k^T)
  }
}

// ---------------- kernels ----------------
// LayerNorm over D=512, one row per 256-thread block (LDS reduction)
__global__ void ln_kernel(const float* __restrict__ x, const float* __restrict__ gamma,
                          const float* __restrict__ beta, float* __restrict__ xnF,
                          __bf16* __restrict__ xnB) {
  const int row = blockIdx.x, t = threadIdx.x;
  const float* xr = x + (size_t)row * D;
  const float a = xr[t], b = xr[t + 256];
  __shared__ float s1[256], s2[256];
  s1[t] = a + b; s2[t] = a * a + b * b;
  __syncthreads();
  for (int off = 128; off > 0; off >>= 1) {
    if (t < off) { s1[t] += s1[t + off]; s2[t] += s2[t + off]; }
    __syncthreads();
  }
  const float mu  = s1[0] * (1.0f / D);
  const float var = s2[0] * (1.0f / D) - mu * mu;
  const float rs  = rsqrtf(var + 1e-5f);
#pragma unroll
  for (int j2 = 0; j2 < 2; ++j2) {
    const int j = t + j2 * 256;
    const float v = (xr[j] - mu) * rs * gamma[j] + beta[j];
    xnF[(size_t)row * D + j] = v;
    xnB[(size_t)row * D + j] = (__bf16)v;
  }
}

__global__ void pack_bf16_kernel(const float* __restrict__ in, __bf16* __restrict__ out, int n) {
  for (int i = blockIdx.x * blockDim.x + threadIdx.x; i < n; i += gridDim.x * blockDim.x)
    out[i] = (__bf16)in[i];
}

// C[m,n] = sum_k A[m,k]*B[n,k] + bias[n]; bf16 out, optional transposed f16 out
__global__ void gemm_nt_bf16_kernel(const __bf16* __restrict__ A, const __bf16* __restrict__ B,
                                    const float* __restrict__ bias, __bf16* __restrict__ outB,
                                    _Float16* __restrict__ outT, int K, int lda, int ldb,
                                    int ldo, int ldt) {
  const int lane = threadIdx.x & 31, wid = threadIdx.x >> 5;
  const int mBase = blockIdx.x * 64 + (wid & 1) * 32;
  const int nBase = blockIdx.y * 128 + (wid >> 1) * 32;
  v8f acc00 = vzero8(), acc01 = vzero8(), acc10 = vzero8(), acc11 = vzero8();
  for (int k = 0; k < K; k += 32) {
    const bf16x16 a0 = load_a_bf16(A, lda, mBase,      k, lane);
    const bf16x16 a1 = load_a_bf16(A, lda, mBase + 16, k, lane);
    const bf16x16 b0 = load_b_bf16(B, ldb, nBase,      k, lane);
    const bf16x16 b1 = load_b_bf16(B, ldb, nBase + 16, k, lane);
    acc00 = wmma_bf16(a0, b0, acc00);
    acc01 = wmma_bf16(a0, b1, acc01);
    acc10 = wmma_bf16(a1, b0, acc10);
    acc11 = wmma_bf16(a1, b1, acc11);
  }
  store_tile(outB, outT, ldo, ldt, mBase,      nBase,      lane, acc00, bias);
  store_tile(outB, outT, ldo, ldt, mBase,      nBase + 16, lane, acc01, bias);
  store_tile(outB, outT, ldo, ldt, mBase + 16, nBase,      lane, acc10, bias);
  store_tile(outB, outT, ldo, ldt, mBase + 16, nBase + 16, lane, acc11, bias);
}

// Pass 1: S = (q k^T)/sqrt(D) via bf16 WMMA.
// All 8 waves of a block share the same m-range -> stage k rows in LDS once per
// block (64KB of the 320KB WGP LDS), cutting L2 B-traffic 8x. q row-block A-frags
// live in registers for the whole kernel (16 frags = full K=512).
__global__ void attn_s_kernel(const __bf16* __restrict__ Q, const __bf16* __restrict__ Kb,
                              _Float16* __restrict__ S, float* __restrict__ pmax,
                              float* __restrict__ psum, int nOff) {
  __shared__ __attribute__((aligned(128))) __bf16 ksh[MSTAGE * D]; // 64KB
  const int tid = threadIdx.x;
  const int lane = tid & 31, wid = tid >> 5;
  const int hi = lane >> 4;
  const int nTileBase = (blockIdx.x * 8 + wid) * 16;     // chunk-local row base
  const int mStart = blockIdx.y * (M / MSPLIT);

  bf16x16 a[16];
#pragma unroll
  for (int t = 0; t < 16; ++t) a[t] = load_a_bf16(Q, D, nOff + nTileBase, t * 32, lane);
  float runMax[8], runSum[8];
#pragma unroll
  for (int r = 0; r < 8; ++r) { runMax[r] = -3.0e38f; runSum[r] = 0.0f; }

  for (int mo = 0; mo < M / MSPLIT; mo += MSTAGE) {
    const int mGlobal = mStart + mo;
    // cooperative stage: MSTAGE rows x 512 cols of k (b128 in, ds b128 out)
    for (int c = tid; c < (MSTAGE * D) / 8; c += 256)
      *(bf16x8*)(ksh + c * 8) = *(const bf16x8*)(Kb + (size_t)mGlobal * D + c * 8);
    __syncthreads();

    for (int mi = 0; mi < MSTAGE; mi += 16) {
      v8f acc = vzero8();
#pragma unroll
      for (int t = 0; t < 16; ++t) {
        const int n   = mi + (lane & 15);
        const int klo = t * 32 + ((lane >> 4) << 4);
        const bf16x16 b = *(const bf16x16*)(ksh + n * D + klo);
        acc = wmma_bf16(a[t], b, acc);
      }
      float s8[8];
#pragma unroll
      for (int r = 0; r < 8; ++r) s8[r] = acc[r] * RSCALE;
      // per-row reductions across the 16-lane half groups (rows hi*8 + r)
#pragma unroll
      for (int r = 0; r < 8; ++r) {
        float v = s8[r];
        for (int off2 = 1; off2 < 16; off2 <<= 1) v = fmaxf(v, __shfl_xor(v, off2, 16));
        const float nm = fmaxf(runMax[r], v);
        const float corr = __expf(runMax[r] - nm);
        float e = __expf(s8[r] - nm);
        for (int off2 = 1; off2 < 16; off2 <<= 1) e += __shfl_xor(e, off2, 16);
        runSum[r] = runSum[r] * corr + e;
        runMax[r] = nm;
        S[(size_t)(nTileBase + hi * 8 + r) * M + (mGlobal + mi) + (lane & 15)] = (_Float16)s8[r];
      }
    }
    __syncthreads();
  }
  if ((lane & 15) == 0) {
#pragma unroll
    for (int r = 0; r < 8; ++r) {
      const int idx = (nTileBase + hi * 8 + r) * MSPLIT + blockIdx.y;
      pmax[idx] = runMax[r];
      psum[idx] = runSum[r];
    }
  }
}

// Combine m-split partial stats: global max + rescaled sum -> inv_sum
__global__ void softmax_stats_kernel(const float* __restrict__ pmax, const float* __restrict__ psum,
                                     float* __restrict__ rowMax, float* __restrict__ rowInv) {
  const int row = blockIdx.x * blockDim.x + threadIdx.x;
  if (row >= NCHUNK) return;
  float gm = -3.0e38f;
#pragma unroll
  for (int i = 0; i < MSPLIT; ++i) gm = fmaxf(gm, pmax[row * MSPLIT + i]);
  float gs = 0.0f;
#pragma unroll
  for (int i = 0; i < MSPLIT; ++i) gs += psum[row * MSPLIT + i] * __expf(pmax[row * MSPLIT + i] - gm);
  rowMax[row] = gm;
  rowInv[row] = 1.0f / gs;
}

// Pass 2: attn = softmax(S) @ k  (NT against k^T, f16 WMMA, exp fused in A-load).
// 32x64 wave tile: 8 wmma per 2 exp-fused A-frags keeps the TRANS-class exp work
// under the WMMA/VALU co-execution ratio and halves k^T traffic per FLOP.
__global__ void attn_av_kernel(const _Float16* __restrict__ S, const _Float16* __restrict__ KT,
                               const float* __restrict__ rowMax, const float* __restrict__ rowInv,
                               __bf16* __restrict__ attnB, int nOff) {
  const int lane = threadIdx.x & 31, wid = threadIdx.x >> 5;
  const int mLocal = blockIdx.x * 128 + (wid & 3) * 32;  // chunk-local q rows
  const int dBase  = blockIdx.y * 128 + (wid >> 2) * 64; // output feature cols
  const int r0 = mLocal + (lane & 15), r1 = r0 + 16;
  const float mx0 = rowMax[r0], is0 = rowInv[r0];
  const float mx1 = rowMax[r1], is1 = rowInv[r1];
  v8f acc[2][4];
#pragma unroll
  for (int i = 0; i < 2; ++i)
#pragma unroll
    for (int j = 0; j < 4; ++j) acc[i][j] = vzero8();

  for (int k = 0; k < M; k += 32) {
    const f16x16 a0 = load_p_f16(S, M, mLocal,      k, lane, mx0, is0);
    const f16x16 a1 = load_p_f16(S, M, mLocal + 16, k, lane, mx1, is1);
#pragma unroll
    for (int j = 0; j < 4; ++j) {
      const f16x16 b = load_b_f16(KT, M, dBase + 16 * j, k, lane);
      acc[0][j] = wmma_f16(a0, b, acc[0][j]);
      acc[1][j] = wmma_f16(a1, b, acc[1][j]);
    }
  }
  const int hi = lane >> 4;
#pragma unroll
  for (int i = 0; i < 2; ++i)
#pragma unroll
    for (int j = 0; j < 4; ++j) {
      const int col = dBase + 16 * j + (lane & 15);
#pragma unroll
      for (int r = 0; r < 8; ++r) {
        const int row = nOff + mLocal + i * 16 + hi * 8 + r;
        attnB[(size_t)row * D + col] = (__bf16)acc[i][j][r];
      }
    }
}

// out = q@WoutA^T + v@WoutB^T + bout + xn  (fused concat-GEMM + bias + residual)
__global__ void gemm_out_kernel(const __bf16* __restrict__ Qb, const __bf16* __restrict__ Vb,
                                const __bf16* __restrict__ Wo, const float* __restrict__ bout,
                                const float* __restrict__ xn, float* __restrict__ out) {
  const int lane = threadIdx.x & 31, wid = threadIdx.x >> 5;
  const int mBase = blockIdx.x * 64 + (wid & 1) * 32;
  const int nBase = blockIdx.y * 128 + (wid >> 1) * 32;
  v8f acc00 = vzero8(), acc01 = vzero8(), acc10 = vzero8(), acc11 = vzero8();
  for (int k = 0; k < D; k += 32) {  // first half of concat: q
    const bf16x16 a0 = load_a_bf16(Qb, D, mBase,      k, lane);
    const bf16x16 a1 = load_a_bf16(Qb, D, mBase + 16, k, lane);
    const bf16x16 b0 = load_b_bf16(Wo, 2 * D, nBase,      k, lane);
    const bf16x16 b1 = load_b_bf16(Wo, 2 * D, nBase + 16, k, lane);
    acc00 = wmma_bf16(a0, b0, acc00); acc01 = wmma_bf16(a0, b1, acc01);
    acc10 = wmma_bf16(a1, b0, acc10); acc11 = wmma_bf16(a1, b1, acc11);
  }
  for (int k = 0; k < D; k += 32) {  // second half of concat: out_v
    const bf16x16 a0 = load_a_bf16(Vb, D, mBase,      k, lane);
    const bf16x16 a1 = load_a_bf16(Vb, D, mBase + 16, k, lane);
    const bf16x16 b0 = load_b_bf16(Wo, 2 * D, nBase,      D + k, lane);
    const bf16x16 b1 = load_b_bf16(Wo, 2 * D, nBase + 16, D + k, lane);
    acc00 = wmma_bf16(a0, b0, acc00); acc01 = wmma_bf16(a0, b1, acc01);
    acc10 = wmma_bf16(a1, b0, acc10); acc11 = wmma_bf16(a1, b1, acc11);
  }
  const int hi = lane >> 4, col = nBase + (lane & 15);
  const float bv0 = bout[col], bv1 = bout[col + 16];
#pragma unroll
  for (int r = 0; r < 8; ++r) {
    const size_t i00 = (size_t)(mBase + hi * 8 + r) * D + col;
    const size_t i10 = (size_t)(mBase + 16 + hi * 8 + r) * D + col;
    out[i00]      = acc00[r] + bv0 + xn[i00];
    out[i00 + 16] = acc01[r] + bv1 + xn[i00 + 16];
    out[i10]      = acc10[r] + bv0 + xn[i10];
    out[i10 + 16] = acc11[r] + bv1 + xn[i10 + 16];
  }
}

// ---------------- host launcher ----------------
extern "C" void kernel_launch(void* const* d_in, const int* in_sizes, int n_in,
                              void* d_out, int out_size, void* d_ws, size_t ws_size,
                              hipStream_t stream) {
  const float* x     = (const float*)d_in[0];
  const float* xhat  = (const float*)d_in[1];
  const float* Wq    = (const float*)d_in[2];
  const float* bq    = (const float*)d_in[3];
  const float* Wk    = (const float*)d_in[4];
  const float* bk    = (const float*)d_in[5];
  const float* Wv    = (const float*)d_in[6];
  const float* bv    = (const float*)d_in[7];
  const float* Wout  = (const float*)d_in[8];
  const float* bout  = (const float*)d_in[9];
  const float* gamma = (const float*)d_in[10];
  const float* beta  = (const float*)d_in[11];
  float* out = (float*)d_out;

  char* w = (char*)d_ws;
  size_t off = 0;
  auto carve = [&](size_t bytes) -> void* {
    void* p = (void*)(w + off);
    off += (bytes + 255) & ~(size_t)255;
    return p;
  };
  float*    xnF   = (float*)   carve((size_t)N * D * 4);
  __bf16*   xnB   = (__bf16*)  carve((size_t)N * D * 2);
  __bf16*   xhatB = (__bf16*)  carve((size_t)M * D * 2);
  __bf16*   qB    = (__bf16*)  carve((size_t)N * D * 2);
  __bf16*   kB    = (__bf16*)  carve((size_t)M * D * 2);
  _Float16* kT    = (_Float16*)carve((size_t)D * M * 2);
  __bf16*   attnB = (__bf16*)  carve((size_t)N * D * 2);
  __bf16*   vB    = (__bf16*)  carve((size_t)N * D * 2);
  __bf16*   WqB   = (__bf16*)  carve((size_t)D * D * 2);
  __bf16*   WkB   = (__bf16*)  carve((size_t)D * D * 2);
  __bf16*   WvB   = (__bf16*)  carve((size_t)D * D * 2);
  __bf16*   WoB   = (__bf16*)  carve((size_t)D * 2 * D * 2);
  float*    pmax  = (float*)   carve((size_t)NCHUNK * MSPLIT * 4);
  float*    psum  = (float*)   carve((size_t)NCHUNK * MSPLIT * 4);
  float*    rMax  = (float*)   carve((size_t)NCHUNK * 4);
  float*    rInv  = (float*)   carve((size_t)NCHUNK * 4);
  _Float16* S     = (_Float16*)carve((size_t)NCHUNK * M * 2);
  (void)ws_size; (void)in_sizes; (void)n_in; (void)out_size;

  const dim3 blk(256);
  ln_kernel<<<N, blk, 0, stream>>>(x, gamma, beta, xnF, xnB);
  pack_bf16_kernel<<<2048, blk, 0, stream>>>(xhat, xhatB, M * D);
  pack_bf16_kernel<<<512,  blk, 0, stream>>>(Wq, WqB, D * D);
  pack_bf16_kernel<<<512,  blk, 0, stream>>>(Wk, WkB, D * D);
  pack_bf16_kernel<<<512,  blk, 0, stream>>>(Wv, WvB, D * D);
  pack_bf16_kernel<<<1024, blk, 0, stream>>>(Wout, WoB, D * 2 * D);

  // projections: q = xn Wq^T + bq ; k = x_hat Wk^T + bk (also emits k^T in f16)
  gemm_nt_bf16_kernel<<<dim3(N / 64, D / 128), blk, 0, stream>>>(xnB,   WqB, bq, qB, nullptr, D, D, D, D, 0);
  gemm_nt_bf16_kernel<<<dim3(M / 64, D / 128), blk, 0, stream>>>(xhatB, WkB, bk, kB, kT,      D, D, D, D, M);

  // attention, chunked so S stays L2-resident
  for (int c = 0; c < N; c += NCHUNK) {
    attn_s_kernel<<<dim3(NCHUNK / 128, MSPLIT), blk, 0, stream>>>(qB, kB, S, pmax, psum, c);
    softmax_stats_kernel<<<NCHUNK / 256, blk, 0, stream>>>(pmax, psum, rMax, rInv);
    attn_av_kernel<<<dim3(NCHUNK / 128, D / 128), blk, 0, stream>>>(S, kT, rMax, rInv, attnB, c);
  }

  // out_v = attn Wv^T + bv ; out = [q,out_v] Wout^T + bout + xn
  gemm_nt_bf16_kernel<<<dim3(N / 64, D / 128), blk, 0, stream>>>(attnB, WvB, bv, vB, nullptr, D, D, D, D, 0);
  gemm_out_kernel<<<dim3(N / 64, D / 128), blk, 0, stream>>>(qB, vB, WoB, bout, xnF, out);
}
